// GRUModel_25615184953905
// MI455X (gfx1250) — matrix operand order repaired and checked by reading, not compiled, not run on hardware
//
#include <hip/hip_runtime.h>
#include <hip/hip_bf16.h>

// ---------------------------------------------------------------------------
// GRU (B=64, T=512, D=256, H=512) for MI455X / gfx1250.
//
//  k_prep       : weights f32 -> bf16 *transposed* ([g][h][k]); zero h0+barrier.
//  k_input_gemm : xg[g] = x @ Wg_x + bg via v_wmma_f32_16x16x32_bf16 (bandwidth
//                 bound: ~224MB streamed, ~10us at 23.3 TB/s).
//  k_scan       : persistent kernel, 64 WGs x 2 waves = 128 waves, one 16x16
//                 tile per wave. Recurrent weights (96KB/WG) are pulled into
//                 LDS ONCE via the Tensor Data Mover (tensor_load_to_lds +
//                 s_wait_tensorcnt), so all 512 timesteps run WMMA entirely
//                 out of LDS. Two grid barriers per step (z/r -> r.*h -> h~).
//  k_tail       : copy outputs[:, T-1, :] to the h_last slot of d_out.
// ---------------------------------------------------------------------------

#define B_   64
#define T_   512
#define D_   256
#define H_   512
#define BT_  (B_ * T_)          // 32768
#define NWG_SCAN 64

typedef __attribute__((ext_vector_type(16))) __bf16   v16bf;
typedef __attribute__((ext_vector_type(8)))  float    v8f;
typedef __attribute__((ext_vector_type(4)))  unsigned u32x4;
typedef __attribute__((ext_vector_type(8)))  unsigned u32x8;

union AFrag { v16bf v; unsigned d[8];  unsigned short u[16]; };
union BFrag { v16bf v; u32x4    q[2];  unsigned short u[16]; };

__device__ __forceinline__ unsigned short f2bf(float f) {
  unsigned u = __float_as_uint(f);
  u += 0x7fffu + ((u >> 16) & 1u);          // round to nearest even
  return (unsigned short)(u >> 16);
}

__device__ __forceinline__ v8f wmma_bf16(v16bf a, v16bf b, v8f c) {
  // (neg_a, A, neg_b, B, c_mod, C, reuse_a, reuse_b)
  return __builtin_amdgcn_wmma_f32_16x16x32_bf16(false, a, false, b,
                                                 (short)0, c, false, false);
}

__device__ __forceinline__ void grid_sync(unsigned* bar) {
  __syncthreads();
  if (threadIdx.x == 0) {
    __threadfence();
    volatile unsigned* vgen = bar + 1;
    unsigned gen = *vgen;
    if (atomicAdd(bar, 1u) == NWG_SCAN - 1) {
      bar[0] = 0;                 // reset arrival counter for next use
      __threadfence();
      atomicAdd(bar + 1, 1u);     // release
    } else {
      while (*vgen == gen) { __builtin_amdgcn_s_sleep(1); }
    }
    __threadfence();
  }
  __syncthreads();
}

// --------------------------- prep ------------------------------------------
__global__ void k_prep(const float* __restrict__ Wz, const float* __restrict__ Wr,
                       const float* __restrict__ Wh,
                       unsigned short* __restrict__ Wxt,   // [3][H][D] bf16
                       unsigned short* __restrict__ Wht,   // [3][H][H] bf16
                       float* __restrict__ h0, unsigned* __restrict__ bar,
                       int total) {
  int i = blockIdx.x * blockDim.x + threadIdx.x;
  if (i < 2) bar[i] = 0u;
  if (i < B_ * H_) h0[i] = 0.0f;
  if (i < total) {                         // total = 3 * 768 * 512
    int g   = i / (768 * H_);
    int rem = i - g * (768 * H_);
    int k   = rem / H_;
    int h   = rem - k * H_;
    const float* W = (g == 0) ? Wz : ((g == 1) ? Wr : Wh);
    unsigned short v = f2bf(W[rem]);
    if (k < D_) Wxt[(size_t)g * H_ * D_ + (size_t)h * D_ + k]        = v;
    else        Wht[(size_t)g * H_ * H_ + (size_t)h * H_ + (k - D_)] = v;
  }
}

// --------------------------- input GEMM ------------------------------------
__global__ __launch_bounds__(256) void k_input_gemm(
    const float* __restrict__ x,                 // [BT][D]
    const unsigned short* __restrict__ Wxt,      // [3][H][D] bf16 transposed
    const float* __restrict__ bz, const float* __restrict__ brr,
    const float* __restrict__ bh,
    float* __restrict__ xg)                      // [3][BT][H]
{
  __shared__ __align__(16) unsigned short sA[16 * D_];   // 8 KB
  const int tid  = threadIdx.x;
  const int lane = tid & 31, wv = tid >> 5;
  const int Wid  = blockIdx.x * 8 + wv;
  const int mt   = Wid >> 5, nt = Wid & 31;      // 2048 x 32 tiles
  const int m0   = mt * 16, n0 = nt * 16;

  for (int i = tid; i < 16 * D_; i += 256) {     // stage x rows as bf16
    int rr = i >> 8, k = i & (D_ - 1);
    sA[i] = f2bf(x[(size_t)(m0 + rr) * D_ + k]);
  }
  __syncthreads();

  const int cn = lane & 15, cmB = (lane >> 4) << 3;
  const int col = n0 + cn;
  const int am = lane & 15, kh = (lane >> 4) << 3, kb = (lane >> 4) << 4;
  const unsigned* sA32 = (const unsigned*)sA;

  v8f acc0, acc1, acc2;
  const float v0 = bz[col], v1 = brr[col], v2 = bh[col];
#pragma unroll
  for (int r = 0; r < 8; ++r) { acc0[r] = v0; acc1[r] = v1; acc2[r] = v2; }

  for (int kk = 0; kk < D_; kk += 32) {
    AFrag a;
#pragma unroll
    for (int j = 0; j < 8; ++j) {
      int kp = kb + (j << 1);
      a.d[j] = sA32[am * (D_ / 2) + ((kk + kp) >> 1)];
    }
    const unsigned short* w0 = Wxt + (size_t)0 * H_ * D_ + (size_t)col * D_ + kk + kb;
    const unsigned short* w1 = Wxt + (size_t)1 * H_ * D_ + (size_t)col * D_ + kk + kb;
    const unsigned short* w2 = Wxt + (size_t)2 * H_ * D_ + (size_t)col * D_ + kk + kb;
    BFrag b0, b1, b2;
    b0.q[0] = ((const u32x4*)w0)[0]; b0.q[1] = ((const u32x4*)w0)[1];
    b1.q[0] = ((const u32x4*)w1)[0]; b1.q[1] = ((const u32x4*)w1)[1];
    b2.q[0] = ((const u32x4*)w2)[0]; b2.q[1] = ((const u32x4*)w2)[1];
    acc0 = wmma_bf16(a.v, b0.v, acc0);
    acc1 = wmma_bf16(a.v, b1.v, acc1);
    acc2 = wmma_bf16(a.v, b2.v, acc2);
  }
#pragma unroll
  for (int r = 0; r < 8; ++r) {
    size_t row = (size_t)(m0 + cmB + r);
    xg[(size_t)0 * BT_ * H_ + row * H_ + col] = acc0[r];
    xg[(size_t)1 * BT_ * H_ + row * H_ + col] = acc1[r];
    xg[(size_t)2 * BT_ * H_ + row * H_ + col] = acc2[r];
  }
}

// --------------------------- recurrent scan --------------------------------
// 64 WGs x 2 waves. Wave Wid = blockIdx*2 + wv owns tile (mt = Wid>>5,
// nt = Wid&31). A WG owns 32 output columns (cb..cb+31); its Wz/Wr/Wh column
// slabs (3 x 32 x 512 bf16 = 96KB) are TDM-loaded into LDS once.
__global__ __launch_bounds__(64) void k_scan(
    const float* __restrict__ xg,                // [3][BT][H]
    const unsigned short* __restrict__ Wht,      // [3][H][H] bf16 transposed
    float* __restrict__ out,                     // [B][T][H]
    const float* __restrict__ h0,                // zeros [B][H]
    unsigned short* __restrict__ rhbuf,          // [B][H] bf16
    unsigned* bar)
{
  __shared__ __align__(16) unsigned short sW[3 * 32 * H_]; // 96 KB weight slabs
  __shared__ __align__(16) unsigned short sA[16 * H_];     // 16 KB staging
  const int tid  = threadIdx.x;
  const int lane = tid & 31, wv = tid >> 5;
  const int Wid  = blockIdx.x * 2 + wv;          // 0..127
  const int mt   = Wid >> 5, nt = Wid & 31;      // 4 x 32 tiles
  const int m0   = mt * 16, n0 = nt * 16;
  const int cb   = (blockIdx.x & 15) * 32;       // first column owned by WG
  const int cn = lane & 15, cmB = (lane >> 4) << 3;
  const int col = n0 + cn;
  const int cl  = col - cb;                      // 0..31 (column within slab)
  const int am = lane & 15, kh = (lane >> 4) << 3, kb = (lane >> 4) << 4;
  const unsigned* sA32 = (const unsigned*)sA;

  // ---- one-time TDM preload of this WG's weight slabs into LDS ----
  // 2D descriptor: 3 rows (gates) of 8192 dwords (32 cols x 512 k bf16),
  // global row stride = H*H/2 dwords. data_size = 4B.
  if (wv == 0) {
    unsigned long long ga = (unsigned long long)(const void*)(Wht + (size_t)cb * H_);
    unsigned lds          = (unsigned)(unsigned long long)(const void*)&sW[0];
    const unsigned tdim0  = 32 * H_ / 2;                       // 8192 dwords/row
    const unsigned tdim1  = 3;
    const unsigned long long str0 = (unsigned long long)H_ * H_ / 2; // 131072
    u32x4 g0; u32x8 g1; u32x4 gz;
    gz[0] = 0u; gz[1] = 0u; gz[2] = 0u; gz[3] = 0u;
    g0[0] = 1u;                                                // count=1 (valid)
    g0[1] = lds;                                               // lds_addr
    g0[2] = (unsigned)ga;                                      // global_addr lo
    g0[3] = (unsigned)((ga >> 32) & 0x1FFFFFFull) | (2u << 30);// addr hi | type=2
    g1[0] = (2u << 16);                                        // data_size=4B
    g1[1] = (tdim0 & 0xFFFFu) << 16;                           // tensor_dim0 lo16
    g1[2] = ((tdim0 >> 16) & 0xFFFFu) | ((tdim1 & 0xFFFFu) << 16);
    g1[3] = ((tdim1 >> 16) & 0xFFFFu) | ((tdim0 & 0xFFFFu) << 16); // tile_dim0
    g1[4] = (tdim1 & 0xFFFFu);                                 // tile_dim1=3
    g1[5] = (unsigned)(str0 & 0xFFFFFFFFull);                  // dim0_stride lo
    g1[6] = (unsigned)((str0 >> 32) & 0xFFFFull);              // dim0_stride hi
    g1[7] = 0u;
    asm volatile("tensor_load_to_lds %0, %1, %2, %3"
                 :: "s"(g0), "s"(g1), "s"(gz), "s"(gz)
                 : "memory");
    __builtin_amdgcn_s_wait_tensorcnt(0);
  }
  __syncthreads();

  for (int t = 0; t < T_; ++t) {
    const float* hbase   = (t == 0) ? h0 : (out + (size_t)(t - 1) * H_);
    const size_t hstride = (t == 0) ? (size_t)H_ : (size_t)T_ * H_;

    // ---- stage h_{t-1} rows m0..m0+15 as bf16 ----
    for (int i = tid; i < 16 * H_; i += 64) {
      int rr = i >> 9, k = i & (H_ - 1);
      sA[i] = f2bf(hbase[(size_t)(m0 + rr) * hstride + k]);
    }
    __syncthreads();

    // ---- phase A: z and r pre-activations (shared A-fragment) ----
    v8f accz, accr;
#pragma unroll
    for (int r = 0; r < 8; ++r) {
      size_t row = (size_t)(m0 + cmB + r);
      accz[r] = xg[(row * T_ + t) * H_ + col];
      accr[r] = xg[(size_t)1 * BT_ * H_ + (row * T_ + t) * H_ + col];
    }
    for (int kk = 0; kk < H_; kk += 32) {
      AFrag a;
#pragma unroll
      for (int j = 0; j < 8; ++j) {
        int kp = kb + (j << 1);
        a.d[j] = sA32[am * (H_ / 2) + ((kk + kp) >> 1)];
      }
      const unsigned short* wz = sW + (size_t)(0 * 32 + cl) * H_ + kk + kb;
      const unsigned short* wr = sW + (size_t)(1 * 32 + cl) * H_ + kk + kb;
      BFrag fz, fr;
      fz.q[0] = ((const u32x4*)wz)[0]; fz.q[1] = ((const u32x4*)wz)[1];
      fr.q[0] = ((const u32x4*)wr)[0]; fr.q[1] = ((const u32x4*)wr)[1];
      accz = wmma_bf16(a.v, fz.v, accz);
      accr = wmma_bf16(a.v, fr.v, accr);
    }
    float zreg[8], hreg[8];
#pragma unroll
    for (int r = 0; r < 8; ++r) {
      int row = m0 + cmB + r;
      float hv = hbase[(size_t)row * hstride + col];
      float z  = 1.0f / (1.0f + __expf(-accz[r]));
      float rg = 1.0f / (1.0f + __expf(-accr[r]));
      zreg[r] = z; hreg[r] = hv;
      rhbuf[row * H_ + col] = f2bf(rg * hv);      // (r .* h) feeds Wh_h matmul
    }
    grid_sync(bar);

    // ---- phase B: h_tilde = leaky_relu(xh + (r.*h) @ Wh_h); blend ----
    for (int i = tid; i < 16 * H_; i += 64) {
      int rr = i >> 9, k = i & (H_ - 1);
      sA[i] = rhbuf[(m0 + rr) * H_ + k];
    }
    __syncthreads();

    v8f acch;
#pragma unroll
    for (int r = 0; r < 8; ++r) {
      size_t row = (size_t)(m0 + cmB + r);
      acch[r] = xg[(size_t)2 * BT_ * H_ + (row * T_ + t) * H_ + col];
    }
    for (int kk = 0; kk < H_; kk += 32) {
      AFrag a;
#pragma unroll
      for (int j = 0; j < 8; ++j) {
        int kp = kb + (j << 1);
        a.d[j] = sA32[am * (H_ / 2) + ((kk + kp) >> 1)];
      }
      const unsigned short* wh = sW + (size_t)(2 * 32 + cl) * H_ + kk + kb;
      BFrag fh;
      fh.q[0] = ((const u32x4*)wh)[0]; fh.q[1] = ((const u32x4*)wh)[1];
      acch = wmma_bf16(a.v, fh.v, acch);
    }
#pragma unroll
    for (int r = 0; r < 8; ++r) {
      float pre = acch[r];
      float htl = pre > 0.0f ? pre : 0.01f * pre;
      float z   = zreg[r];
      size_t row = (size_t)(m0 + cmB + r);
      out[(row * T_ + t) * H_ + col] = (1.0f - z) * hreg[r] + z * htl;
    }
    grid_sync(bar);
  }
}

// --------------------------- h_last tail -----------------------------------
__global__ void k_tail(const float* __restrict__ out, float* __restrict__ hl) {
  int i = blockIdx.x * blockDim.x + threadIdx.x;
  if (i < B_ * H_) {
    int b = i >> 9, h = i & (H_ - 1);
    hl[i] = out[((size_t)b * T_ + (T_ - 1)) * H_ + h];
  }
}

// --------------------------- launch ----------------------------------------
extern "C" void kernel_launch(void* const* d_in, const int* in_sizes, int n_in,
                              void* d_out, int out_size, void* d_ws, size_t ws_size,
                              hipStream_t stream) {
  const float* x  = (const float*)d_in[0];
  const float* Wz = (const float*)d_in[1];
  const float* bz = (const float*)d_in[2];
  const float* Wr = (const float*)d_in[3];
  const float* br = (const float*)d_in[4];
  const float* Wh = (const float*)d_in[5];
  const float* bh = (const float*)d_in[6];
  float* out = (float*)d_out;

  // workspace layout (256B aligned offsets)
  char* ws = (char*)d_ws;
  const size_t OFF_XG  = 0;                                   // 3*BT*H f32 = 192 MB
  const size_t OFF_WX  = OFF_XG + (size_t)3 * BT_ * H_ * 4;   // 3*H*D bf16
  const size_t OFF_WH  = OFF_WX + (size_t)3 * H_ * D_ * 2;    // 3*H*H bf16
  const size_t OFF_RH  = OFF_WH + (size_t)3 * H_ * H_ * 2;    // B*H bf16
  const size_t OFF_H0  = OFF_RH + (size_t)B_ * H_ * 2;        // B*H f32 zeros
  const size_t OFF_BAR = OFF_H0 + (size_t)B_ * H_ * 4;        // 2 u32

  float*          xg    = (float*)(ws + OFF_XG);
  unsigned short* Wxt   = (unsigned short*)(ws + OFF_WX);
  unsigned short* Wht   = (unsigned short*)(ws + OFF_WH);
  unsigned short* rhbuf = (unsigned short*)(ws + OFF_RH);
  float*          h0    = (float*)(ws + OFF_H0);
  unsigned*       bar   = (unsigned*)(ws + OFF_BAR);

  const int prepN = 3 * (D_ + H_) * H_;                       // 1,179,648
  k_prep<<<(prepN + 255) / 256, 256, 0, stream>>>(Wz, Wr, Wh, Wxt, Wht, h0, bar, prepN);

  const int gemmBlocks = (BT_ / 16) * (H_ / 16) / 8;          // 8192
  k_input_gemm<<<gemmBlocks, 256, 0, stream>>>(x, Wxt, bz, br, bh, xg);

  k_scan<<<NWG_SCAN, 64, 0, stream>>>(xg, Wht, out, h0, rhbuf, bar);

  k_tail<<<(B_ * H_ + 255) / 256, 256, 0, stream>>>(out, out + (size_t)BT_ * H_);
}
